// SlotAttention_70738111365382
// MI455X (gfx1250) — compile-verified, matrix operand order checked by reading, not run on hardware
//
#include <hip/hip_runtime.h>
#include <hip/hip_bf16.h>

// ---------------- CDNA5 (gfx1250) Slot Attention, bf16 WMMA pipeline --------
// wave32; v_wmma_f32_16x16x32_bf16 for all GEMMs; f32 for LN/softmax/gates.
// TDM (tensor_load_to_lds) stages the attn matrix into LDS for the
// updates GEMM; gemm_kv register-blocks 8 column tiles to cut x re-reads.

typedef __bf16 bf16_t;
typedef __attribute__((ext_vector_type(16))) __bf16 v16bf;
typedef __attribute__((ext_vector_type(8)))  float  v8f;
typedef __attribute__((ext_vector_type(4)))  unsigned int u32x4;
typedef __attribute__((ext_vector_type(8)))  int i32x8;
typedef __attribute__((ext_vector_type(4)))  int i32x4;

static constexpr int B_  = 64;
static constexpr int N_  = 4096;
static constexpr int D_  = 256;
static constexpr int K_  = 15;   // real slots
static constexpr int KS_ = 16;   // padded slots (WMMA M)
static constexpr int H_  = 512;
static constexpr int HN_ = 2048; // half of N staged in LDS at a time (64KB)

#if defined(__AMDGCN__) && __has_builtin(__builtin_amdgcn_tensor_load_to_lds)
#define HAS_TDM 1
#else
#define HAS_TDM 0
#endif

// ---- fragment loader: 16-bit A/B layout (ISA 7.12.2) ----
// lane = row/col index (lane&15); hi = lane>>4 selects K halves.
// halves j=0..7  -> K = kBase + 8*hi + j          (contiguous)
// halves j=8..15 -> K = kBase + 16 + 8*hi + (j-8) (contiguous)
__device__ __forceinline__ v16bf load_frag(const bf16_t* __restrict__ base,
                                           int ld, int row, int kBase, int hi) {
    const bf16_t* p = base + (size_t)row * ld + kBase + 8 * hi;
    union { v16bf v; uint4 u[2]; } f;
    f.u[0] = *(const uint4*)(p);
    f.u[1] = *(const uint4*)(p + 16);
    return f.v;
}

__device__ __forceinline__ v8f wmma_bf16(v16bf a, v16bf b, v8f c) {
    return __builtin_amdgcn_wmma_f32_16x16x32_bf16(
        /*neg_a=*/false, a, /*neg_b=*/false, b,
        /*c_mod=*/(short)0, c, /*reuse_a=*/false, /*reuse_b=*/false);
}

#if HAS_TDM
// Issue a 2D TDM load: tile (tile_d0 x tile_d1) of 2-byte elements from a
// row-major tensor with row stride stride_d0 (elements) into LDS at lds_off.
// D# layout per CDNA5 ISA ch.8 (group0 128b, group1 256b).
__device__ __forceinline__ void tdm_load_2d(unsigned lds_off, const void* gaddr,
                                            unsigned tensor_d0, unsigned tensor_d1,
                                            unsigned tile_d0, unsigned tile_d1,
                                            unsigned stride_d0) {
    unsigned long long ga = (unsigned long long)(uintptr_t)gaddr;
    u32x4 g0;
    g0[0] = 1u;                                 // count=1 (valid), user mode
    g0[1] = lds_off;                            // lds_addr (bytes)
    g0[2] = (unsigned)ga;                       // global_addr[31:0]
    g0[3] = (unsigned)(ga >> 32) | (2u << 30);  // global_addr[56:32] | type=2
    i32x8 g1;
    g1[0] = (int)(1u << 16);                    // data_size=1 (2 bytes)
    g1[1] = (int)(tensor_d0 << 16);             // tensor_dim0[15:0]
    g1[2] = (int)((tensor_d0 >> 16) | (tensor_d1 << 16));
    g1[3] = (int)((tensor_d1 >> 16) | (tile_d0 << 16));
    g1[4] = (int)(tile_d1 & 0xFFFFu);           // tile_dim1 | tile_dim2=0
    g1[5] = (int)stride_d0;                     // tensor_dim0_stride[31:0]
    g1[6] = 0;
    g1[7] = 0;
    i32x4 z4 = {0, 0, 0, 0};
#if __clang_major__ >= 23
    i32x8 z8 = {0, 0, 0, 0, 0, 0, 0, 0};
    __builtin_amdgcn_tensor_load_to_lds(g0, g1, z4, z4, z8, 0);
#else
    __builtin_amdgcn_tensor_load_to_lds(g0, g1, z4, z4, 0);
#endif
}
#endif

// ---------------- elementwise helpers ----------------
__global__ void cvt_f32_bf16_kernel(const float* __restrict__ in,
                                    bf16_t* __restrict__ out, int n) {
    int i = blockIdx.x * blockDim.x + threadIdx.x;
    if (i < n) out[i] = (bf16_t)in[i];
}

__global__ void zero_f32_kernel(float* __restrict__ p, int n) {
    int i = blockIdx.x * blockDim.x + threadIdx.x;
    if (i < n) p[i] = 0.0f;
}

// slots = mu + exp(logsigma)*noise, padded to KS rows (row 15 = 0)
__global__ void slots_init_kernel(const float* __restrict__ mu,
                                  const float* __restrict__ logsig,
                                  const float* __restrict__ noise,
                                  float* __restrict__ slots) {
    int i = blockIdx.x * blockDim.x + threadIdx.x;
    int total = B_ * KS_ * D_;
    if (i >= total) return;
    int row = i / D_, d = i % D_;
    int b = row / KS_, kk = row % KS_;
    float v = 0.0f;
    if (kk < K_) {
        int wi = kk * D_ + d;
        v = mu[wi] + __expf(logsig[wi]) * noise[(b * K_ + kk) * D_ + d];
    }
    slots[i] = v;
}

// LayerNorm over rows of length 256; one wave per row; output bf16.
__global__ void ln_rows_kernel(const float* __restrict__ in,
                               bf16_t* __restrict__ out,
                               const float* __restrict__ w,
                               const float* __restrict__ bvec, int rows) {
    int wave = threadIdx.x >> 5, lane = threadIdx.x & 31;
    int row = blockIdx.x * 8 + wave;
    if (row >= rows) return;
    const float* p = in + (size_t)row * D_ + lane * 8;
    float4 a0 = *(const float4*)p;
    float4 a1 = *(const float4*)(p + 4);
    float x[8] = {a0.x, a0.y, a0.z, a0.w, a1.x, a1.y, a1.z, a1.w};
    float s = 0.f;
    for (int j = 0; j < 8; ++j) s += x[j];
    for (int off = 1; off < 32; off <<= 1) s += __shfl_xor(s, off, 32);
    float m = s * (1.0f / D_);
    float vs = 0.f;
    for (int j = 0; j < 8; ++j) { float d = x[j] - m; vs += d * d; }
    for (int off = 1; off < 32; off <<= 1) vs += __shfl_xor(vs, off, 32);
    float rstd = rsqrtf(vs * (1.0f / D_) + 1e-5f);
    const float* wp = w + lane * 8;
    const float* bp = bvec + lane * 8;
    union { uint4 u; bf16_t h[8]; } pk;
    for (int j = 0; j < 8; ++j)
        pk.h[j] = (bf16_t)((x[j] - m) * rstd * wp[j] + bp[j]);
    *(uint4*)(out + (size_t)row * D_ + lane * 8) = pk.u;
}

// ---------------- big fused GEMM: k = x@Wk^T (row-major), vT = (x@Wv^T)^T ---
// Register-blocked: one wave computes 16 rows x 8 column tiles, reusing the
// A fragment across all 8 WMMAs -> x is streamed only 4x total.
__global__ void gemm_kv_kernel(const bf16_t* __restrict__ x,
                               const bf16_t* __restrict__ Wk,
                               const bf16_t* __restrict__ Wv,
                               bf16_t* __restrict__ kOut,
                               bf16_t* __restrict__ vT) {
    int wave = threadIdx.x >> 5, lane = threadIdx.x & 31;
    int mTile = blockIdx.x * 8 + wave;         // 16384 row tiles
    int which  = blockIdx.y >> 1;              // 0 = k, 1 = v
    int cGroup = blockIdx.y & 1;               // 8 col tiles per group
    int hi = lane >> 4, ln = lane & 15;
    int rowA = mTile * 16 + ln;
    const bf16_t* W = which ? Wv : Wk;
    v8f acc[8] = {};
    for (int kb = 0; kb < D_; kb += 32) {
        v16bf a = load_frag(x, D_, rowA, kb, hi);
#pragma unroll
        for (int c = 0; c < 8; ++c) {
            int col = (cGroup * 8 + c) * 16 + ln;
            v16bf b = load_frag(W, D_, col, kb, hi);
            acc[c] = wmma_bf16(a, b, acc[c]);
        }
    }
    if (which == 0) {
#pragma unroll
        for (int c = 0; c < 8; ++c) {
            int col = (cGroup * 8 + c) * 16 + ln;
            for (int r = 0; r < 8; ++r) {
                int rowg = mTile * 16 + 8 * hi + r;
                kOut[(size_t)rowg * D_ + col] = (bf16_t)acc[c][r];
            }
        }
    } else {
        // transposed store: vT[b][d=col][n]; 8 consecutive n per lane -> uint4
        int rowg0 = mTile * 16 + 8 * hi;
        int b = rowg0 >> 12;                   // N_ = 4096
        int n0 = rowg0 & (N_ - 1);
#pragma unroll
        for (int c = 0; c < 8; ++c) {
            int col = (cGroup * 8 + c) * 16 + ln;
            union { uint4 u; bf16_t h[8]; } pk;
            for (int r = 0; r < 8; ++r) pk.h[r] = (bf16_t)acc[c][r];
            *(uint4*)(vT + (size_t)b * D_ * N_ + (size_t)col * N_ + n0) = pk.u;
        }
    }
}

// ---------------- generic small GEMM: out = act(A@W^T * scale + bias + resid)
__global__ void wmma_gemm_kernel(const bf16_t* __restrict__ A,
                                 const bf16_t* __restrict__ Bw,
                                 const float* __restrict__ bias,
                                 const float* __restrict__ resid,
                                 float* __restrict__ outF,
                                 bf16_t* __restrict__ outB,
                                 int Kd, int Ncols, float scale, int relu) {
    int wave = threadIdx.x >> 5, lane = threadIdx.x & 31;
    int nColTiles = Ncols >> 4;
    int tile = blockIdx.x * 8 + wave;
    int mTile = tile / nColTiles, cTile = tile % nColTiles;
    int hi = lane >> 4, ln = lane & 15;
    int rowA = mTile * 16 + ln;
    int col  = cTile * 16 + ln;
    v8f acc = {};
    for (int kb = 0; kb < Kd; kb += 32) {
        v16bf a = load_frag(A, Kd, rowA, kb, hi);
        v16bf b = load_frag(Bw, Kd, col, kb, hi);
        acc = wmma_bf16(a, b, acc);
    }
    float bcol = bias ? bias[col] : 0.0f;
    for (int r = 0; r < 8; ++r) {
        int rowg = mTile * 16 + 8 * hi + r;
        float v = acc[r] * scale + bcol;
        if (resid) v += resid[(size_t)rowg * Ncols + col];
        if (relu) v = fmaxf(v, 0.0f);
        if (outF) outF[(size_t)rowg * Ncols + col] = v;
        if (outB) outB[(size_t)rowg * Ncols + col] = (bf16_t)v;
    }
}

// ---------------- attention: logits tile + fused softmax over slots ---------
// One wave handles ALL 16 (padded) slots x 16 tokens. Softmax over M.
__global__ void attn_kernel(const bf16_t* __restrict__ q,
                            const bf16_t* __restrict__ kmat,
                            bf16_t* __restrict__ attn_bf,
                            float* __restrict__ sums,
                            float* __restrict__ attn_out_f32, int writeF32) {
    int wave = threadIdx.x >> 5, lane = threadIdx.x & 31;
    int b = blockIdx.y;
    int nTile = blockIdx.x * 8 + wave;
    int hi = lane >> 4, ln = lane & 15;
    const bf16_t* qb = q    + (size_t)b * KS_ * D_;
    const bf16_t* kb = kmat + (size_t)b * N_  * D_;
    int colN = nTile * 16 + ln;
    v8f acc = {};
    for (int kd = 0; kd < D_; kd += 32) {
        v16bf a = load_frag(qb, D_, ln, kd, hi);
        v16bf bb = load_frag(kb, D_, colN, kd, hi);
        acc = wmma_bf16(a, bb, acc);
    }
    if (hi) acc[7] = -3.0e38f;   // mask pad slot m=15
    // softmax over M (8 regs + lane^16 partner shares same column n)
    float mx = acc[0];
    for (int r = 1; r < 8; ++r) mx = fmaxf(mx, acc[r]);
    mx = fmaxf(mx, __shfl_xor(mx, 16, 32));
    float e[8], s = 0.f;
    for (int r = 0; r < 8; ++r) { e[r] = __expf(acc[r] - mx); s += e[r]; }
    s += __shfl_xor(s, 16, 32);
    float inv = 1.0f / s;
    for (int r = 0; r < 8; ++r) {
        float p = e[r] * inv;
        int m = r + 8 * hi;
        attn_bf[(size_t)b * KS_ * N_ + (size_t)m * N_ + colN] = (bf16_t)p;
        if (writeF32) {
            if (m < K_)
                attn_out_f32[((size_t)b * K_ + m) * N_ + colN] = p;
        } else {
            float t = p;                       // reduce over 16 tokens in tile
            for (int off = 1; off < 16; off <<= 1) t += __shfl_xor(t, off, 32);
            if (ln == 0) atomicAdd(&sums[b * KS_ + m], t);
        }
    }
}

// ---------------- updates = (attn / (sum+eps)) @ v ----------------
// A matrix (16 x 4096 bf16, 128KB) is shared by all 8 waves of the block:
// stage it into LDS in two 64KB halves via the Tensor Data Mover.
__global__ void updates_kernel(const bf16_t* __restrict__ attn_bf,
                               const bf16_t* __restrict__ vT,
                               const float* __restrict__ sums,
                               float* __restrict__ updF,
                               bf16_t* __restrict__ updB) {
    __shared__ bf16_t asl[KS_ * HN_];          // 16 rows x 2048 cols = 64KB
    int wave = threadIdx.x >> 5, lane = threadIdx.x & 31;
    int b = blockIdx.y;
    int cTile = blockIdx.x * 8 + wave;         // 16 column (d) tiles total
    int hi = lane >> 4, ln = lane & 15;
    const bf16_t* Ab = attn_bf + (size_t)b * KS_ * N_;
    const bf16_t* Vb = vT      + (size_t)b * D_  * N_;
    int col = cTile * 16 + ln;
    v8f acc = {};
    for (int h = 0; h < 2; ++h) {
        __syncthreads();                       // previous half fully consumed
#if HAS_TDM
        if (threadIdx.x < 32) {
            tdm_load_2d((unsigned)(uintptr_t)(void*)asl, Ab + h * HN_,
                        /*tensor_d0=*/N_, /*tensor_d1=*/KS_,
                        /*tile_d0=*/HN_, /*tile_d1=*/KS_,
                        /*stride_d0=*/N_);
            __builtin_amdgcn_s_wait_tensorcnt(0);
        }
#else
        for (int i = threadIdx.x; i < KS_ * HN_ / 8; i += 256) {
            int r = i >> 8, cidx = i & 255;    // 256 uint4 per row
            ((uint4*)asl)[i] =
                *(const uint4*)(Ab + (size_t)r * N_ + h * HN_ + cidx * 8);
        }
#endif
        __syncthreads();
        for (int kb = 0; kb < HN_; kb += 32) {
            v16bf a = load_frag(asl, HN_, ln, kb, hi);
            v16bf bb = load_frag(Vb, N_, col, h * HN_ + kb, hi);
            acc = wmma_bf16(a, bb, acc);
        }
    }
    for (int r = 0; r < 8; ++r) {
        int m = r + 8 * hi;
        float invs = 1.0f / (sums[b * KS_ + m] + 1e-8f);
        float v = acc[r] * invs;
        size_t idx = ((size_t)b * KS_ + m) * D_ + col;
        updF[idx] = v;
        updB[idx] = (bf16_t)v;
    }
}

// ---------------- GRU gate math (elementwise) ----------------
__global__ void gru_gate_kernel(const float* __restrict__ gi,
                                const float* __restrict__ gh,
                                float* __restrict__ slots) {
    int i = blockIdx.x * blockDim.x + threadIdx.x;
    int total = B_ * KS_ * D_;
    if (i >= total) return;
    int row = i / D_, d = i % D_;
    const float* gir = gi + (size_t)row * 3 * D_;
    const float* ghr = gh + (size_t)row * 3 * D_;
    float ir = gir[d], iz = gir[D_ + d], inn = gir[2 * D_ + d];
    float hr = ghr[d], hz = ghr[D_ + d], hn  = ghr[2 * D_ + d];
    float r = 1.0f / (1.0f + __expf(-(ir + hr)));
    float z = 1.0f / (1.0f + __expf(-(iz + hz)));
    float n = tanhf(inn + r * hn);
    float h = slots[i];
    slots[i] = (1.0f - z) * n + z * h;
}

__global__ void copy_slots_kernel(const float* __restrict__ slots,
                                  float* __restrict__ out) {
    int i = blockIdx.x * blockDim.x + threadIdx.x;
    int total = B_ * K_ * D_;
    if (i >= total) return;
    int row = i / D_, d = i % D_;
    int b = row / K_, kk = row % K_;
    out[i] = slots[((size_t)b * KS_ + kk) * D_ + d];
}

// ---------------- host launch ----------------
extern "C" void kernel_launch(void* const* d_in, const int* in_sizes, int n_in,
                              void* d_out, int out_size, void* d_ws, size_t ws_size,
                              hipStream_t stream) {
    const float* inputs   = (const float*)d_in[0];
    const float* noise    = (const float*)d_in[1];
    const float* mu       = (const float*)d_in[2];
    const float* logsig   = (const float*)d_in[3];
    const float* Wq       = (const float*)d_in[4];
    const float* Wk       = (const float*)d_in[5];
    const float* Wv       = (const float*)d_in[6];
    const float* w_ih     = (const float*)d_in[7];
    const float* w_hh     = (const float*)d_in[8];
    const float* b_ih     = (const float*)d_in[9];
    const float* b_hh     = (const float*)d_in[10];
    const float* mlp_w1   = (const float*)d_in[11];
    const float* mlp_b1   = (const float*)d_in[12];
    const float* mlp_w2   = (const float*)d_in[13];
    const float* mlp_b2   = (const float*)d_in[14];
    const float* ln_in_w  = (const float*)d_in[15];
    const float* ln_in_b  = (const float*)d_in[16];
    const float* ln_sl_w  = (const float*)d_in[17];
    const float* ln_sl_b  = (const float*)d_in[18];
    const float* ln_mlp_w = (const float*)d_in[19];
    const float* ln_mlp_b = (const float*)d_in[20];

    float* out_slots = (float*)d_out;                       // (B,15,256)
    float* out_attn  = out_slots + (size_t)B_ * K_ * D_;    // (B,15,4096)

    // workspace carve (256B aligned)
    char* wsp = (char*)d_ws;
    auto alloc = [&](size_t bytes) -> void* {
        void* p = (void*)wsp;
        wsp += (bytes + 255) & ~(size_t)255;
        return p;
    };
    bf16_t* x_bf   = (bf16_t*)alloc((size_t)B_ * N_ * D_ * 2);
    bf16_t* k_bf   = (bf16_t*)alloc((size_t)B_ * N_ * D_ * 2);
    bf16_t* vT_bf  = (bf16_t*)alloc((size_t)B_ * N_ * D_ * 2);
    bf16_t* attn_bf= (bf16_t*)alloc((size_t)B_ * KS_ * N_ * 2);
    bf16_t* q_bf   = (bf16_t*)alloc((size_t)B_ * KS_ * D_ * 2);
    bf16_t* s_bf   = (bf16_t*)alloc((size_t)B_ * KS_ * D_ * 2);
    bf16_t* s2_bf  = (bf16_t*)alloc((size_t)B_ * KS_ * D_ * 2);
    bf16_t* sr_bf  = (bf16_t*)alloc((size_t)B_ * KS_ * D_ * 2);
    bf16_t* upd_bf = (bf16_t*)alloc((size_t)B_ * KS_ * D_ * 2);
    bf16_t* h1_bf  = (bf16_t*)alloc((size_t)B_ * KS_ * H_ * 2);
    float*  slots_f= (float*) alloc((size_t)B_ * KS_ * D_ * 4);
    float*  upd_f  = (float*) alloc((size_t)B_ * KS_ * D_ * 4);
    float*  gi_f   = (float*) alloc((size_t)B_ * KS_ * 3 * D_ * 4);
    float*  gh_f   = (float*) alloc((size_t)B_ * KS_ * 3 * D_ * 4);
    float*  sums_f = (float*) alloc((size_t)B_ * KS_ * 4);
    bf16_t* wq_bf  = (bf16_t*)alloc((size_t)D_ * D_ * 2);
    bf16_t* wk_bf  = (bf16_t*)alloc((size_t)D_ * D_ * 2);
    bf16_t* wv_bf  = (bf16_t*)alloc((size_t)D_ * D_ * 2);
    bf16_t* wih_bf = (bf16_t*)alloc((size_t)3 * D_ * D_ * 2);
    bf16_t* whh_bf = (bf16_t*)alloc((size_t)3 * D_ * D_ * 2);
    bf16_t* w1_bf  = (bf16_t*)alloc((size_t)H_ * D_ * 2);
    bf16_t* w2_bf  = (bf16_t*)alloc((size_t)D_ * H_ * 2);

    const int TB = 256;
    auto cvt = [&](const float* in, bf16_t* out, int n) {
        cvt_f32_bf16_kernel<<<(n + TB - 1) / TB, TB, 0, stream>>>(in, out, n);
    };
    // weights -> bf16
    cvt(Wq, wq_bf, D_ * D_);
    cvt(Wk, wk_bf, D_ * D_);
    cvt(Wv, wv_bf, D_ * D_);
    cvt(w_ih, wih_bf, 3 * D_ * D_);
    cvt(w_hh, whh_bf, 3 * D_ * D_);
    cvt(mlp_w1, w1_bf, H_ * D_);
    cvt(mlp_w2, w2_bf, D_ * H_);

    // x = LN(inputs) -> bf16
    ln_rows_kernel<<<(B_ * N_) / 8, TB, 0, stream>>>(inputs, x_bf, ln_in_w, ln_in_b, B_ * N_);
    // k, vT (register-blocked: blockIdx.y = {k,v} x {2 col groups})
    gemm_kv_kernel<<<dim3((B_ * N_ / 16) / 8, 4), TB, 0, stream>>>(x_bf, wk_bf, wv_bf, k_bf, vT_bf);
    // slot init
    slots_init_kernel<<<(B_ * KS_ * D_) / TB, TB, 0, stream>>>(mu, logsig, noise, slots_f);

    const float scale = 0.0625f;   // D^-0.5

    for (int t = 0; t < 4; ++t) {
        // s = LN(slots); q = s @ Wq^T * scale
        ln_rows_kernel<<<(B_ * KS_) / 8, TB, 0, stream>>>(slots_f, s_bf, ln_sl_w, ln_sl_b, B_ * KS_);
        wmma_gemm_kernel<<<((B_ * KS_ / 16) * (D_ / 16)) / 8, TB, 0, stream>>>(
            s_bf, wq_bf, nullptr, nullptr, nullptr, q_bf, D_, D_, scale, 0);

        if (t < 3) {
            zero_f32_kernel<<<(B_ * KS_ + TB - 1) / TB, TB, 0, stream>>>(sums_f, B_ * KS_);
            attn_kernel<<<dim3((N_ / 16) / 8, B_), TB, 0, stream>>>(
                q_bf, k_bf, attn_bf, sums_f, nullptr, 0);
            updates_kernel<<<dim3((D_ / 16) / 8, B_), TB, 0, stream>>>(
                attn_bf, vT_bf, sums_f, upd_f, upd_bf);
            // GRU: gi = upd @ w_ih^T + b_ih ; gh = slots @ w_hh^T + b_hh
            cvt_f32_bf16_kernel<<<(B_ * KS_ * D_) / TB, TB, 0, stream>>>(
                slots_f, sr_bf, B_ * KS_ * D_);
            wmma_gemm_kernel<<<((B_ * KS_ / 16) * (3 * D_ / 16)) / 8, TB, 0, stream>>>(
                upd_bf, wih_bf, b_ih, nullptr, gi_f, nullptr, D_, 3 * D_, 1.0f, 0);
            wmma_gemm_kernel<<<((B_ * KS_ / 16) * (3 * D_ / 16)) / 8, TB, 0, stream>>>(
                sr_bf, whh_bf, b_hh, nullptr, gh_f, nullptr, D_, 3 * D_, 1.0f, 0);
            gru_gate_kernel<<<(B_ * KS_ * D_) / TB, TB, 0, stream>>>(gi_f, gh_f, slots_f);
            // MLP: slots += relu(LN(slots) @ w1^T + b1) @ w2^T + b2
            ln_rows_kernel<<<(B_ * KS_) / 8, TB, 0, stream>>>(slots_f, s2_bf, ln_mlp_w, ln_mlp_b, B_ * KS_);
            wmma_gemm_kernel<<<((B_ * KS_ / 16) * (H_ / 16)) / 8, TB, 0, stream>>>(
                s2_bf, w1_bf, mlp_b1, nullptr, nullptr, h1_bf, D_, H_, 1.0f, 1);
            wmma_gemm_kernel<<<((B_ * KS_ / 16) * (D_ / 16)) / 8, TB, 0, stream>>>(
                h1_bf, w2_bf, mlp_b2, slots_f, slots_f, nullptr, H_, D_, 1.0f, 0);
        } else {
            // final attention -> f32 output (no /sum normalization)
            attn_kernel<<<dim3((N_ / 16) / 8, B_), TB, 0, stream>>>(
                q_bf, k_bf, attn_bf, sums_f, out_attn, 1);
            copy_slots_kernel<<<(B_ * K_ * D_ + TB - 1) / TB, TB, 0, stream>>>(slots_f, out_slots);
        }
    }
}
